// GuidedCrossAttention_22093311771379
// MI455X (gfx1250) — matrix operand order repaired and checked by reading, not compiled, not run on hardware
//
#include <hip/hip_runtime.h>
#include <hip/hip_bf16.h>

#define NQ 4096
#define NK 4096
#define QD 256
#define KD 320
#define HID 256
#define NH 8
#define DH 32
#define NB 16
#define ATT_SCALE 0.17677669529663687f /* 1/sqrt(32) */

typedef __bf16 bf16;
typedef __attribute__((ext_vector_type(16))) __bf16 v16bf;
typedef __attribute__((ext_vector_type(8)))  float  v8f;

union FragU { v16bf v; uint4 u[2]; bf16 h[16]; };

// ---- WMMA fragment helpers (contiguous K = 16*hi + j for both A and B, wave32) ----

// A fragment 16x32 bf16: lane l in [0,16): row=l, elems j -> K j (0..15)
//                        lane l in [16,32): row=l-16, elems j -> K 16+j (16..31)
__device__ __forceinline__ v16bf load_frag_a(const bf16* base, int ld, int lane) {
  const int r = lane & 15, hi = lane >> 4;
  const bf16* p = base + r * ld + hi * 16;
  FragU t;
  t.u[0] = *(const uint4*)(p);        // K = 16*hi .. 16*hi+7
  t.u[1] = *(const uint4*)(p + 8);    // K = 16*hi+8 .. 16*hi+15
  return t.v;
}

// B fragment 32x16 bf16 where `base` is laid out [N][K] (elements contiguous along K):
// lane l: N = l&15, elems j -> K = j + 16*(l>=16)
__device__ __forceinline__ v16bf load_frag_b_rows(const bf16* base, long ld, int lane) {
  const int n = lane & 15, hi = lane >> 4;
  const bf16* p = base + n * ld + hi * 16;
  FragU t;
  t.u[0] = *(const uint4*)(p);
  t.u[1] = *(const uint4*)(p + 8);
  return t.v;
}

__device__ __forceinline__ v8f wmma_bf16(v16bf a, v16bf b, v8f c) {
  return __builtin_amdgcn_wmma_f32_16x16x32_bf16(false, a, false, b, (short)0, c,
                                                 false, false);
}

// ---- 16-lane row reductions: DPP row_ror rotate-combine (VALU, no LDS) ----
#if __has_builtin(__builtin_amdgcn_update_dpp)
#define ROW_ROR_F32(x, N) \
  __int_as_float(__builtin_amdgcn_update_dpp(0, __float_as_int(x), 0x120 + (N), 0xf, 0xf, true))
#else
#define ROW_ROR_F32(x, N) __shfl_xor((x), (N), 32)
#endif

__device__ __forceinline__ float red16_max(float x) {
  x = fmaxf(x, ROW_ROR_F32(x, 1));
  x = fmaxf(x, ROW_ROR_F32(x, 2));
  x = fmaxf(x, ROW_ROR_F32(x, 4));
  x = fmaxf(x, ROW_ROR_F32(x, 8));
  return x;
}
__device__ __forceinline__ float red16_sum(float x) {
  x += ROW_ROR_F32(x, 1);
  x += ROW_ROR_F32(x, 2);
  x += ROW_ROR_F32(x, 4);
  x += ROW_ROR_F32(x, 8);
  return x;
}

// ---------------- prep kernels (tiny, VALU) ----------------

__global__ __launch_bounds__(256) void fuse_qkv_kernel(
    const float* __restrict__ W, const float* __restrict__ bvec,
    const float* __restrict__ inW, const float* __restrict__ inB,
    bf16* __restrict__ WE, float* __restrict__ bE)
{
  const int a = blockIdx.x, o = threadIdx.x;
  const float* wr = W + (long)a * HID;
  const float* ir = inW + (long)o * HID;
  float s = 0.f;
  for (int m = 0; m < HID; ++m) s += wr[m] * ir[m];
  WE[(long)a * HID + o] = (bf16)s;
  if (a == 0) {
    float sb = inB[o];
    for (int m = 0; m < HID; ++m) sb += bvec[m] * ir[m];
    bE[o] = sb;
  }
}

__global__ __launch_bounds__(256) void fuse_out_kernel(
    const float* __restrict__ mow, const float* __restrict__ mob,
    const float* __restrict__ Wo, const float* __restrict__ bo,
    bf16* __restrict__ WE, float* __restrict__ bE)
{
  const int h = blockIdx.x, q = threadIdx.x;
  float s = 0.f;
  for (int m = 0; m < HID; ++m) s += mow[(long)h * HID + m] * Wo[(long)m * QD + q];
  WE[(long)h * QD + q] = (bf16)s;
  if (h == 0) {
    float sb = bo[q];
    for (int m = 0; m < HID; ++m) sb += mob[m] * Wo[(long)m * QD + q];
    bE[q] = sb;
  }
}

__global__ void cast_kernel(const float* __restrict__ in, bf16* __restrict__ out, int n) {
  const int i = blockIdx.x * blockDim.x + threadIdx.x;
  if (i < n) out[i] = (bf16)in[i];
}

// per-batch contiguous key range (batch indices are sorted)
__global__ void batch_range_kernel(const int* __restrict__ kbat,
                                   int* __restrict__ kstart, int* __restrict__ kend) {
  const int b = threadIdx.x;
  if (b >= NB) return;
  int lo = 0, hi = NK;
  while (lo < hi) { int mid = (lo + hi) >> 1; if (kbat[mid] <  b) lo = mid + 1; else hi = mid; }
  kstart[b] = lo;
  lo = 0; hi = NK;
  while (lo < hi) { int mid = (lo + hi) >> 1; if (kbat[mid] <= b) lo = mid + 1; else hi = mid; }
  kend[b] = lo;
}

// ---------------- bf16 WMMA GEMM: C[M,N] = A[M,K] @ B[K,N] + bias ----------------
// OUT_MODE: 0 = f32 row-major, 1 = bf16 row-major, 2 = bf16 transposed [N][ldc]
template <int OUT_MODE>
__global__ __launch_bounds__(128) void gemm_bf16_kernel(
    const bf16* __restrict__ A, int lda,
    const bf16* __restrict__ B, int ldb,
    const float* __restrict__ bias,
    void* __restrict__ Cout, int ldc, int K)
{
  __shared__ __align__(16) bf16 As[64 * 32];  // [m][k], pitch 32
  __shared__ __align__(16) bf16 Bs[64 * 40];  // transposed [n][k], pitch 40
  const int tid = threadIdx.x;
  const int lane = tid & 31, w = tid >> 5;
  const int wm = w >> 1, wn = w & 1;
  const long m_blk = (long)blockIdx.x * 64;
  const int  n_blk = blockIdx.y * 64;

  v8f acc[2][2];
#pragma unroll
  for (int i = 0; i < 2; ++i)
#pragma unroll
    for (int j = 0; j < 2; ++j) acc[i][j] = (v8f){0,0,0,0,0,0,0,0};

  for (int k0 = 0; k0 < K; k0 += 32) {
    __syncthreads();
    {
      const int row = tid >> 1, col = (tid & 1) * 16;
      const bf16* src = A + (m_blk + row) * (long)lda + k0 + col;
      *(uint4*)(As + row * 32 + col)     = *(const uint4*)(src);
      *(uint4*)(As + row * 32 + col + 8) = *(const uint4*)(src + 8);
    }
    {
      const int kk = tid >> 2, n0 = (tid & 3) * 16;
      const bf16* src = B + (long)(k0 + kk) * ldb + n_blk + n0;
      FragU t;
      t.u[0] = *(const uint4*)(src);
      t.u[1] = *(const uint4*)(src + 8);
#pragma unroll
      for (int j = 0; j < 16; ++j) Bs[(n0 + j) * 40 + kk] = t.h[j];
    }
    __syncthreads();

    v16bf af[2], bfr[2];
    af[0]  = load_frag_a(As + (wm * 32 + 0)  * 32, 32, lane);
    af[1]  = load_frag_a(As + (wm * 32 + 16) * 32, 32, lane);
    bfr[0] = load_frag_b_rows(Bs + (wn * 32 + 0)  * 40, 40, lane);
    bfr[1] = load_frag_b_rows(Bs + (wn * 32 + 16) * 40, 40, lane);
#pragma unroll
    for (int mi = 0; mi < 2; ++mi)
#pragma unroll
      for (int ni = 0; ni < 2; ++ni)
        acc[mi][ni] = wmma_bf16(af[mi], bfr[ni], acc[mi][ni]);
  }

  const int hi = lane >> 4, ln = lane & 15;
#pragma unroll
  for (int mi = 0; mi < 2; ++mi)
#pragma unroll
    for (int ni = 0; ni < 2; ++ni) {
      const long row0 = m_blk + wm * 32 + mi * 16 + hi * 8;
      const int  col  = n_blk + wn * 32 + ni * 16 + ln;
      const float bv = bias[col];
      if (OUT_MODE == 2) {
        FragU t;
#pragma unroll
        for (int r = 0; r < 8; ++r) t.h[r] = (bf16)(acc[mi][ni][r] + bv);
        *(uint4*)((bf16*)Cout + (long)col * ldc + row0) = t.u[0];
      } else {
#pragma unroll
        for (int r = 0; r < 8; ++r) {
          const float vv = acc[mi][ni][r] + bv;
          if (OUT_MODE == 1) ((bf16*)Cout)[(row0 + r) * ldc + col] = (bf16)vv;
          else               ((float*)Cout)[(row0 + r) * ldc + col] = vv;
        }
      }
    }
}

// ---------------- flash attention, one wave per (32-q tile, head) ----------------
// Block-diagonal pruning: sorted batch ids -> contiguous valid key range per q-tile.
__global__ __launch_bounds__(128) void attn_kernel(
    const bf16* __restrict__ Qm, const bf16* __restrict__ Km, const bf16* __restrict__ Vt,
    const int* __restrict__ qbat, const int* __restrict__ kbat,
    const int* __restrict__ kstart, const int* __restrict__ kend,
    bf16* __restrict__ ctx)
{
  __shared__ __align__(16) bf16 Pl[4][2][16 * 32];  // per-wave, per-qtile P staging
  const int lane = threadIdx.x & 31, w = threadIdx.x >> 5;
  const int tile = blockIdx.x * 4 + w;              // (NQ/32)*NH tiles
  const int head = tile & (NH - 1);
  const int q0   = (tile >> 3) * 32;                // 32 q rows per wave
  const int hi = lane >> 4, ln = lane & 15;
  const int hoff = head * DH;

  v16bf qa[2];
  qa[0] = load_frag_a(Qm + (long)(q0 + 0)  * HID + hoff, HID, lane);
  qa[1] = load_frag_a(Qm + (long)(q0 + 16) * HID + hoff, HID, lane);
  int qb[2][8];
#pragma unroll
  for (int t = 0; t < 2; ++t)
#pragma unroll
    for (int r = 0; r < 8; ++r) qb[t][r] = qbat[q0 + t * 16 + r + 8 * hi];

  // contiguous valid key range for this q-tile
  const int bmin = qbat[q0], bmax = qbat[q0 + 31];
  const int kt0 = kstart[bmin] & ~31;
  const int kt1 = (kend[bmax] + 31) & ~31;

  float mrun[2][8], lrun[2][8];
#pragma unroll
  for (int t = 0; t < 2; ++t)
#pragma unroll
    for (int r = 0; r < 8; ++r) { mrun[t][r] = -1e30f; lrun[t][r] = 0.0f; }
  v8f acc[2][2] = {{{}, {}}, {{}, {}}};
  const bf16* vt0 = Vt + (long)(hoff + 0)  * NK;
  const bf16* vt1 = Vt + (long)(hoff + 16) * NK;

  for (int kt = kt0; kt < kt1; kt += 32) {
    const v16bf kf0 = load_frag_b_rows(Km + (long)kt * HID + hoff, HID, lane);
    const v16bf kf1 = load_frag_b_rows(Km + (long)(kt + 16) * HID + hoff, HID, lane);
    const int kb0 = kbat[kt + ln];
    const int kb1 = kbat[kt + 16 + ln];
    const v8f z = {};

#pragma unroll
    for (int t = 0; t < 2; ++t) {
      v8f s0 = wmma_bf16(qa[t], kf0, z);
      v8f s1 = wmma_bf16(qa[t], kf1, z);
      float p0[8], p1[8];
#pragma unroll
      for (int r = 0; r < 8; ++r) {
        p0[r] = (qb[t][r] == kb0) ? s0[r] * ATT_SCALE : -1e30f;
        p1[r] = (qb[t][r] == kb1) ? s1[r] * ATT_SCALE : -1e30f;
      }
#pragma unroll
      for (int r = 0; r < 8; ++r) {
        const float mx   = red16_max(fmaxf(p0[r], p1[r]));
        const float mnew = fmaxf(mrun[t][r], mx);
        const float alpha = __expf(mrun[t][r] - mnew);
        mrun[t][r] = mnew;
        p0[r] = __expf(p0[r] - mnew);
        p1[r] = __expf(p1[r] - mnew);
        const float rs = red16_sum(p0[r] + p1[r]);
        lrun[t][r] = lrun[t][r] * alpha + rs;
        acc[t][0][r] *= alpha;
        acc[t][1][r] *= alpha;
      }
      bf16* pw = Pl[w][t];
#pragma unroll
      for (int r = 0; r < 8; ++r) {
        pw[(r + 8 * hi) * 32 + ln]      = (bf16)p0[r];
        pw[(r + 8 * hi) * 32 + 16 + ln] = (bf16)p1[r];
      }
    }
    asm volatile("s_wait_dscnt 0" ::: "memory");
    __builtin_amdgcn_wave_barrier();

    const v16bf v0 = load_frag_b_rows(vt0 + kt, NK, lane);
    const v16bf v1 = load_frag_b_rows(vt1 + kt, NK, lane);
#pragma unroll
    for (int t = 0; t < 2; ++t) {
      const v16bf pa = load_frag_a(Pl[w][t], 32, lane);
      acc[t][0] = wmma_bf16(pa, v0, acc[t][0]);
      acc[t][1] = wmma_bf16(pa, v1, acc[t][1]);
    }

    if (kt + 32 < kt1) {
      __builtin_prefetch(Km + (long)(kt + 32) * HID + hoff, 0, 1);
      __builtin_prefetch(vt0 + kt + 32, 0, 1);
      __builtin_prefetch(vt1 + kt + 32, 0, 1);
    }
    __builtin_amdgcn_wave_barrier();
  }

#pragma unroll
  for (int t = 0; t < 2; ++t)
#pragma unroll
    for (int r = 0; r < 8; ++r) {
      const float inv = (lrun[t][r] > 0.0f) ? 1.0f / lrun[t][r] : 0.0f;
      const long row = q0 + t * 16 + r + 8 * hi;
      ctx[row * HID + hoff + ln]      = (bf16)(acc[t][0][r] * inv);
      ctx[row * HID + hoff + 16 + ln] = (bf16)(acc[t][1][r] * inv);
    }
}

// ---------------- residual + LayerNorm ----------------
__global__ __launch_bounds__(256) void resid_ln_kernel(
    const float* __restrict__ x, const float* __restrict__ y,
    const float* __restrict__ g, const float* __restrict__ b,
    float* __restrict__ out)
{
  __shared__ float red[256];
  const int row = blockIdx.x, t = threadIdx.x;
  const float v = x[(long)row * QD + t] + y[(long)row * QD + t];
  red[t] = v; __syncthreads();
  for (int s = 128; s > 0; s >>= 1) { if (t < s) red[t] += red[t + s]; __syncthreads(); }
  const float mu = red[0] * (1.0f / QD);
  __syncthreads();
  const float d = v - mu;
  red[t] = d * d; __syncthreads();
  for (int s = 128; s > 0; s >>= 1) { if (t < s) red[t] += red[t + s]; __syncthreads(); }
  const float var = red[0] * (1.0f / QD);
  const float rstd = rsqrtf(var + 1e-5f);
  out[(long)row * QD + t] = d * rstd * g[t] + b[t];
}

// ---------------- host glue ----------------
extern "C" void kernel_launch(void* const* d_in, const int* in_sizes, int n_in,
                              void* d_out, int out_size, void* d_ws, size_t ws_size,
                              hipStream_t stream)
{
  (void)in_sizes; (void)n_in; (void)out_size; (void)ws_size;
  const float* query_nodes = (const float*)d_in[0];
  const float* key_nodes   = (const float*)d_in[1];
  const float* Wq  = (const float*)d_in[2];
  const float* bq  = (const float*)d_in[3];
  const float* Wk  = (const float*)d_in[4];
  const float* bk  = (const float*)d_in[5];
  const float* Wv  = (const float*)d_in[6];
  const float* bv  = (const float*)d_in[7];
  const float* ipw = (const float*)d_in[8];
  const float* ipb = (const float*)d_in[9];
  const float* mow = (const float*)d_in[10];
  const float* mob = (const float*)d_in[11];
  const float* Wo  = (const float*)d_in[12];
  const float* bo  = (const float*)d_in[13];
  const float* lng = (const float*)d_in[14];
  const float* lnb = (const float*)d_in[15];
  const int* qbat = (const int*)d_in[16];
  const int* kbat = (const int*)d_in[17];

  char* ws = (char*)d_ws;
  size_t off = 0;
  auto alloc = [&](size_t bytes) -> void* {
    void* p = ws + off;
    off = (off + bytes + 255) & ~(size_t)255;
    return p;
  };
  bf16* WqE = (bf16*)alloc((size_t)QD * HID * 2);
  bf16* WkE = (bf16*)alloc((size_t)KD * HID * 2);
  bf16* WvE = (bf16*)alloc((size_t)KD * HID * 2);
  bf16* WoE = (bf16*)alloc((size_t)HID * QD * 2);
  float* bqE = (float*)alloc(HID * 4);
  float* bkE = (float*)alloc(HID * 4);
  float* bvE = (float*)alloc(HID * 4);
  float* boE = (float*)alloc(QD * 4);
  int* kst = (int*)alloc(NB * 4);
  int* ken = (int*)alloc(NB * 4);
  bf16* Xq = (bf16*)alloc((size_t)NQ * QD * 2);
  bf16* Xk = (bf16*)alloc((size_t)NK * KD * 2);
  bf16* Qb = (bf16*)alloc((size_t)NQ * HID * 2);
  bf16* Kb = (bf16*)alloc((size_t)NK * HID * 2);
  bf16* Vt = (bf16*)alloc((size_t)HID * NK * 2);   // V transposed [HID][NK]
  bf16* Cb = (bf16*)alloc((size_t)NQ * HID * 2);
  float* Of = (float*)alloc((size_t)NQ * QD * 4);

  fuse_qkv_kernel<<<QD, 256, 0, stream>>>(Wq, bq, ipw,                 ipb,           WqE, bqE);
  fuse_qkv_kernel<<<KD, 256, 0, stream>>>(Wk, bk, ipw + HID * HID,     ipb + HID,     WkE, bkE);
  fuse_qkv_kernel<<<KD, 256, 0, stream>>>(Wv, bv, ipw + 2 * HID * HID, ipb + 2 * HID, WvE, bvE);
  fuse_out_kernel<<<HID, 256, 0, stream>>>(mow, mob, Wo, bo, WoE, boE);
  batch_range_kernel<<<1, 32, 0, stream>>>(kbat, kst, ken);

  cast_kernel<<<(NQ * QD) / 256, 256, 0, stream>>>(query_nodes, Xq, NQ * QD);
  cast_kernel<<<(NK * KD) / 256, 256, 0, stream>>>(key_nodes,   Xk, NK * KD);

  dim3 gqkv(NQ / 64, HID / 64);
  gemm_bf16_kernel<1><<<gqkv, 128, 0, stream>>>(Xq, QD, WqE, HID, bqE, Qb, HID, QD);
  gemm_bf16_kernel<1><<<gqkv, 128, 0, stream>>>(Xk, KD, WkE, HID, bkE, Kb, HID, KD);
  gemm_bf16_kernel<2><<<gqkv, 128, 0, stream>>>(Xk, KD, WvE, HID, bvE, Vt, NK,  KD);

  // flash attention with block-diagonal pruning: (NQ/32)*NH waves, 4 per block
  attn_kernel<<<(NQ / 32) * NH / 4, 128, 0, stream>>>(Qb, Kb, Vt, qbat, kbat, kst, ken, Cb);

  gemm_bf16_kernel<0><<<dim3(NQ / 64, QD / 64), 128, 0, stream>>>(Cb, HID, WoE, QD, boE, Of, QD, HID);
  resid_ln_kernel<<<NQ, 256, 0, stream>>>(query_nodes, Of, lng, lnb, (float*)d_out);
}
